// GRU_58634893525361
// MI455X (gfx1250) — matrix operand order repaired and checked by reading, not compiled, not run on hardware
//
#include <hip/hip_runtime.h>
#include <hip/hip_bf16.h>

typedef __attribute__((ext_vector_type(16))) __bf16 v16bf;
typedef __attribute__((ext_vector_type(8)))  float  v8f;

constexpr int kH = 1024;
constexpr int kD = 96;      // 3*NB
constexpr int kB = 512;
constexpr int kTOUT = 25;

// ---------------------------------------------------------------------------
// CDNA5 async global->LDS copies (ASYNCcnt-tracked, §10 / §15.18.3).
// LDS destination: per-lane 32-bit LDS byte address (addrspace(3) pointer);
// global source: 64-bit address in a VGPR pair.
// ---------------------------------------------------------------------------
__device__ __forceinline__ void async_b128(void* lds_dst, const void* gsrc) {
  asm volatile("global_load_async_to_lds_b128 %0, %1, off"
               :
               : "v"((__attribute__((address_space(3))) void*)lds_dst), "v"(gsrc)
               : "memory");
}
__device__ __forceinline__ void async_b64(void* lds_dst, const void* gsrc) {
  asm volatile("global_load_async_to_lds_b64 %0, %1, off"
               :
               : "v"((__attribute__((address_space(3))) void*)lds_dst), "v"(gsrc)
               : "memory");
}
#define ASYNC_WAIT(n) asm volatile("s_wait_asynccnt " #n ::: "memory")

// ---------------------------------------------------------------------------
// Fragment load from LDS: two 16-byte chunks -> v16bf (2x ds_load_b128)
// ---------------------------------------------------------------------------
__device__ __forceinline__ v16bf ldfrag(const __bf16* p0, const __bf16* p1) {
  union U { uint4 q[2]; v16bf v; } u;
  u.q[0] = *(const uint4*)p0;
  u.q[1] = *(const uint4*)p1;
  return u.v;
}

__device__ __forceinline__ float sigmoidf_(float x) {
  return 1.f / (1.f + __expf(-x));
}

// One K=32 chunk of the 3-gate GEMM. NTGT: accumulator slot for the n-gate
// (3 = ghn during the h-phase, 2 = gin during the x-phase).
template<int NTGT>
__device__ __forceinline__ void gemm_chunk(const __bf16 (*lA)[40], const __bf16 (*lB)[40],
                                           v8f (&acc)[4][2], int wm, int wn, int lane) {
  const int hi = lane >> 4;
  const int lm = lane & 15;
  const int arow = wm * 16 + lm;
  const int kb  = hi * 8;
  v16bf a = ldfrag(&lA[arow][kb], &lA[arow][kb + 16]);
#pragma unroll
  for (int g = 0; g < 3; ++g) {
#pragma unroll
    for (int f = 0; f < 2; ++f) {
      const int brow = g * 64 + wn * 32 + f * 16 + lm;
      const int kb2 = hi * 8;
      v16bf b = ldfrag(&lB[brow][kb2], &lB[brow][kb2 + 16]);
      const int tgt = (g == 2) ? NTGT : g;
      acc[tgt][f] = __builtin_amdgcn_wmma_f32_16x16x32_bf16(
          false, a, false, b, (short)0, acc[tgt][f], false, false);
    }
  }
}

// ---------------------------------------------------------------------------
// Fused GRU step, async-pipelined with per-lane pointer walking.
// Operands pre-converted to bf16: x_bf (B x 96), hprev_bf (B x H),
// Whh (3H x H), Wih (3H x 96).  hprev_f32 feeds the exact z*h term; the
// epilogue writes hnew as f32 + bf16 shadow (next step's A matrix).
// grid = (H/64, B/64), block = 256 (8 waves).  K chunks: 32 of h + 3 of x.
// Per wave per chunk: 4 async b128 ops -> steady-state s_wait_asynccnt 4.
// ---------------------------------------------------------------------------
__global__ __launch_bounds__(256) void gru_step_kernel(
    const __bf16* __restrict__ xbf,
    const __bf16* __restrict__ hprev_bf,
    const float*  __restrict__ hprev,
    float*        __restrict__ hnew,
    __bf16*       __restrict__ hnew_bf,
    const __bf16* __restrict__ Whh,
    const __bf16* __restrict__ Wih,
    const float*  __restrict__ bih,
    const float*  __restrict__ bhh) {
  __shared__ __bf16 lA[2][64][40];    // double-buffered, padded rows (80 B)
  __shared__ __bf16 lB[2][192][40];

  const int tid  = threadIdx.x;
  const int lane = tid & 31;
  const int wave = tid >> 5;
  const int wm = wave >> 1;
  const int wn = wave & 1;
  const int m0 = blockIdx.y * 64;
  const int j0 = blockIdx.x * 64;

  // fixed per-lane staging slots
  const int arow = tid >> 2;          // A: 64 rows x 4 16B-quads
  const int aq   = tid & 3;
  int bg[3], brow[3], bq[3];
#pragma unroll
  for (int i = 0; i < 3; ++i) {
    int cc = tid + i * 256;
    bg[i] = cc >> 8; brow[i] = (cc >> 2) & 63; bq[i] = cc & 3;
  }

  // LDS destinations for both buffers, precomputed
  void* dA[2];
  void* dB[2][3];
#pragma unroll
  for (int bf = 0; bf < 2; ++bf) {
    dA[bf] = &lA[bf][arow][aq * 8];
#pragma unroll
    for (int i = 0; i < 3; ++i) dB[bf][i] = &lB[bf][bg[i] * 64 + brow[i]][bq[i] * 8];
  }

  // walking global source pointers (advance 32 bf16 = 64 B per chunk)
  const __bf16* aP = hprev_bf + (size_t)(m0 + arow) * kH + aq * 8;
  const __bf16* bP[3];
#pragma unroll
  for (int i = 0; i < 3; ++i)
    bP[i] = Whh + (size_t)(bg[i] * kH + j0 + brow[i]) * kH + bq[i] * 8;

  auto stageCur = [&](int bf) {
    async_b128(dA[bf], aP);
#pragma unroll
    for (int i = 0; i < 3; ++i) async_b128(dB[bf][i], bP[i]);
    aP += 32;
#pragma unroll
    for (int i = 0; i < 3; ++i) bP[i] += 32;
  };

  v8f vz = {};
  v8f acc[4][2];
#pragma unroll
  for (int g = 0; g < 4; ++g)
#pragma unroll
    for (int f = 0; f < 2; ++f) acc[g][f] = vz;

  stageCur(0);                                      // chunk 0 -> buf 0
  // ---- h phase: chunks 0..31 ----
#pragma unroll 2
  for (int c = 0; c < 32; ++c) {
    const int buf = c & 1;
    if (c == 31) {                                  // switch sources to x phase
      aP = xbf + (size_t)(m0 + arow) * kD + aq * 8;
#pragma unroll
      for (int i = 0; i < 3; ++i)
        bP[i] = Wih + (size_t)(bg[i] * kH + j0 + brow[i]) * kD + bq[i] * 8;
    }
    stageCur(buf ^ 1);                              // chunk c+1
    ASYNC_WAIT(4);                                  // in-order: chunk c done
    __syncthreads();
    gemm_chunk<3>(lA[buf], lB[buf], acc, wm, wn, lane);
    __syncthreads();
  }
  // ---- x phase: chunks 32..34 ----
#pragma unroll
  for (int c = 32; c < 35; ++c) {
    const int buf = c & 1;
    if (c < 34) { stageCur(buf ^ 1); ASYNC_WAIT(4); }
    else        { ASYNC_WAIT(0); }
    __syncthreads();
    gemm_chunk<2>(lA[buf], lB[buf], acc, wm, wn, lane);
    __syncthreads();
  }

  // ---- fused gate epilogue ----
  const int hi = lane >> 4;
  const int lm = lane & 15;
#pragma unroll
  for (int f = 0; f < 2; ++f) {
    const int j = j0 + wn * 32 + f * 16 + lm;
    const float br   = bih[j] + bhh[j];
    const float bz   = bih[kH + j] + bhh[kH + j];
    const float bin_ = bih[2 * kH + j];
    const float bhn_ = bhh[2 * kH + j];
#pragma unroll
    for (int i = 0; i < 8; ++i) {
      const int b = m0 + wm * 16 + hi * 8 + i;      // C/D layout: M = i + 8*hi
      float r = sigmoidf_(acc[0][f][i] + br);
      float z = sigmoidf_(acc[1][f][i] + bz);
      float n = tanhf(acc[2][f][i] + bin_ + r * (acc[3][f][i] + bhn_));
      float hp = hprev[(size_t)b * kH + j];
      float hv = (1.f - z) * n + z * hp;
      hnew[(size_t)b * kH + j]    = hv;
      hnew_bf[(size_t)b * kH + j] = (__bf16)hv;
    }
  }
}

// ---------------------------------------------------------------------------
// Output projection: out[b,d] = h[b,:].WoutT[d,:] + bout[d], plus compact
// bf16 copy (next GRU step's x).  Async double-buffered, pointer-walked.
// grid = (D/32, B/64), block = 256. Per wave per chunk: 1 b128 + 1 b64.
// ---------------------------------------------------------------------------
__global__ __launch_bounds__(256) void proj_kernel(
    const __bf16* __restrict__ hbf,       // (B, H) bf16
    const __bf16* __restrict__ WoutT,     // (D, H) bf16
    const float*  __restrict__ bout,
    float*        __restrict__ out, int ostride,
    __bf16*       __restrict__ xnext) {   // (B, D) bf16 compact
  __shared__ __bf16 lA[2][64][40];
  __shared__ __bf16 lB[2][32][40];
  const int tid  = threadIdx.x;
  const int lane = tid & 31;
  const int wave = tid >> 5;
  const int wm = wave >> 1;
  const int wn = wave & 1;
  const int m0 = blockIdx.y * 64;
  const int d0 = blockIdx.x * 32;

  const int arow = tid >> 2, aq = tid & 3;      // A: 64 rows x 4 16B-quads
  const int br_  = tid >> 3, bq = tid & 7;      // B: 32 rows x 8 b64-slots

  void* dA[2] = { &lA[0][arow][aq * 8], &lA[1][arow][aq * 8] };
  void* dB[2] = { &lB[0][br_][bq * 4],  &lB[1][br_][bq * 4] };
  const __bf16* aP = hbf   + (size_t)(m0 + arow) * kH + aq * 8;
  const __bf16* bP = WoutT + (size_t)(d0 + br_) * kH + bq * 4;

  auto stageCur = [&](int bf) {
    async_b128(dA[bf], aP);
    async_b64(dB[bf], bP);
    aP += 32; bP += 32;
  };

  v8f acc = {};
  stageCur(0);
#pragma unroll 2
  for (int c = 0; c < 32; ++c) {
    const int buf = c & 1;
    if (c < 31) { stageCur(buf ^ 1); ASYNC_WAIT(2); }
    else        { ASYNC_WAIT(0); }
    __syncthreads();
    const int hi = lane >> 4, lm = lane & 15;
    v16bf a = ldfrag(&lA[buf][wm * 16 + lm][hi * 8],
                     &lA[buf][wm * 16 + lm][hi * 8 + 16]);
    v16bf b = ldfrag(&lB[buf][wn * 16 + lm][hi * 8],
                     &lB[buf][wn * 16 + lm][hi * 8 + 16]);
    acc = __builtin_amdgcn_wmma_f32_16x16x32_bf16(false, a, false, b, (short)0, acc,
                                                  false, false);
    __syncthreads();
  }
  const int hi = lane >> 4, lm = lane & 15;
  const int d = d0 + wn * 16 + lm;
  const float bo = bout[d];
#pragma unroll
  for (int i = 0; i < 8; ++i) {
    const int b = m0 + wm * 16 + hi * 8 + i;
    float v = acc[i] + bo;
    out[(size_t)b * ostride + d] = v;
    xnext[(size_t)b * kD + d] = (__bf16)v;
  }
}

// ---------------------------------------------------------------------------
// Utility kernels
// ---------------------------------------------------------------------------
__global__ void k_cvt_bf16(const float* __restrict__ s, __bf16* __restrict__ d, int n4) {
  int i = blockIdx.x * blockDim.x + threadIdx.x;
  if (i < n4) {
    float4 v = ((const float4*)s)[i];
    __bf16* p = d + (size_t)i * 4;
    p[0] = (__bf16)v.x; p[1] = (__bf16)v.y; p[2] = (__bf16)v.z; p[3] = (__bf16)v.w;
  }
}

__global__ void k_cvt_transpose(const float* __restrict__ s /* H x D */,
                                __bf16* __restrict__ d /* D x H */) {
  int i = blockIdx.x * blockDim.x + threadIdx.x;
  if (i < kH * kD) {
    int k = i / kD, c = i % kD;
    d[(size_t)c * kH + k] = (__bf16)s[i];
  }
}

__global__ void k_zero(float* p, int n) {
  int i = blockIdx.x * blockDim.x + threadIdx.x;
  if (i < n) p[i] = 0.f;
}

// ---------------------------------------------------------------------------
// Host driver.  The reference's encoder scan is dead code (enc_state unused),
// so only the 25 decoder steps run.  Workspace ~13.5 MB.
// ---------------------------------------------------------------------------
extern "C" void kernel_launch(void* const* d_in, const int* in_sizes, int n_in,
                              void* d_out, int out_size, void* d_ws, size_t ws_size,
                              hipStream_t stream) {
  (void)in_sizes; (void)n_in; (void)out_size; (void)ws_size;
  const float* dec_in = (const float*)d_in[1];
  const float* dWih   = (const float*)d_in[6];
  const float* dWhh   = (const float*)d_in[7];
  const float* dbih   = (const float*)d_in[8];
  const float* dbhh   = (const float*)d_in[9];
  const float* Wout   = (const float*)d_in[10];
  const float* bout   = (const float*)d_in[11];
  float* out = (float*)d_out;

  char* w = (char*)d_ws;
  size_t o = 0;
  __bf16* wWhh   = (__bf16*)(w + o); o += (size_t)3 * kH * kH * 2;  // 6291456
  __bf16* wWih   = (__bf16*)(w + o); o += (size_t)3 * kH * kD * 2;  // 589824
  __bf16* wWoutT = (__bf16*)(w + o); o += (size_t)kD * kH * 2;      // 196608
  float*  hA     = (float*)(w + o);  o += (size_t)kB * kH * 4;      // 2097152
  float*  hB     = (float*)(w + o);  o += (size_t)kB * kH * 4;
  __bf16* hAbf   = (__bf16*)(w + o); o += (size_t)kB * kH * 2;      // 1048576
  __bf16* hBbf   = (__bf16*)(w + o); o += (size_t)kB * kH * 2;
  __bf16* xbf    = (__bf16*)(w + o); o += (size_t)kB * kD * 2;      // 98304

  { int n4 = 3 * kH * kH / 4;
    k_cvt_bf16<<<(n4 + 255) / 256, 256, 0, stream>>>(dWhh, wWhh, n4); }
  { int n4 = 3 * kH * kD / 4;
    k_cvt_bf16<<<(n4 + 255) / 256, 256, 0, stream>>>(dWih, wWih, n4); }
  { int n = kH * kD;
    k_cvt_transpose<<<(n + 255) / 256, 256, 0, stream>>>(Wout, wWoutT); }
  { int n4 = kB * kD / 4;                        // x0 = decoder_inputs -> bf16
    k_cvt_bf16<<<(n4 + 255) / 256, 256, 0, stream>>>(dec_in, xbf, n4); }
  { int n = kB * kH;
    k_zero<<<(n + 255) / 256, 256, 0, stream>>>(hA, n); }
  { int n = kB * kH / 2;                         // zero bf16 h shadow
    k_zero<<<(n + 255) / 256, 256, 0, stream>>>((float*)hAbf, n); }

  dim3 gGru(kH / 64, kB / 64);   // (16, 8)
  dim3 gProj(kD / 32, kB / 64);  // (3, 8)
  float*  hc = hA;   float*  hn = hB;
  __bf16* hcb = hAbf; __bf16* hnb = hBbf;
  for (int t = 0; t < kTOUT; ++t) {
    gru_step_kernel<<<gGru, 256, 0, stream>>>(xbf, hcb, hc, hn, hnb,
                                              wWhh, wWih, dbih, dbhh);
    proj_kernel<<<gProj, 256, 0, stream>>>(hnb, wWoutT, bout,
                                           out + (size_t)t * kD, kTOUT * kD, xbf);
    { float*  tf = hc;  hc  = hn;  hn  = tf; }
    { __bf16* tb = hcb; hcb = hnb; hnb = tb; }
  }
}
